// MultiHeadDotProductAttention_28604482191541
// MI455X (gfx1250) — matrix operand order repaired and checked by reading, not compiled, not run on hardware
//
#include <hip/hip_runtime.h>
#include <hip/hip_bf16.h>

// ---------------------------------------------------------------------------
// MultiHeadDotProductAttention forward for MI455X (gfx1250, wave32, WMMA).
// Pipeline: [QKV bf16 WMMA GEMMs] -> [flash-attention WMMA kernel] -> [out GEMM]
// All matmuls use v_wmma_f32_16x16x32_bf16 (bf16 inputs, f32 accumulate).
// Round 3: explicit double-buffered fragment registers in the attention
// QK/PV stages so LDS fragment loads pipeline ahead of the WMMAs instead of
// serializing on a recycled register pair (seen as s_wait_dscnt 0x0 chains).
// ---------------------------------------------------------------------------

typedef __bf16 bf16;
typedef __bf16 v16bf __attribute__((ext_vector_type(16)));
typedef float  v8f   __attribute__((ext_vector_type(8)));

#define DEV_INLINE __device__ __forceinline__

constexpr int Bsz = 2;
constexpr int Seq = 2048;
constexpr int Emb = 2048;
constexpr int Hn  = 16;
constexpr int Dh  = 128;
constexpr int Mtok = Bsz * Seq;           // 4096 tokens

DEV_INLINE v8f wmma_bf16(v16bf a, v16bf b, v8f c) {
  // D = A(16x32 bf16) * B(32x16 bf16) + C(16x16 f32)
  return __builtin_amdgcn_wmma_f32_16x16x32_bf16(
      /*neg_a=*/false, a, /*neg_b=*/false, b,
      /*c_mod=*/(short)0, c, /*reuse_a=*/false, /*reuse_b=*/false);
}

// ---------------------------------------------------------------------------
// Tiled GEMM:  C[M,N] = A[M,K] * W[K,N] (+ bias[n])
//   MODE 0: A is f32, output bf16 scattered to [B,H,S,D] layout, scaled.
//   MODE 1: A is bf16, output f32 row-major [M,N] (final projection).
// Block: 128x128 tile, 256 threads = 8 waves in a 4(M) x 2(N) grid,
// each wave owns 32x64 -> 2x4 WMMA accumulators. K step = 32.
// Software pipeline: global->VGPR staging for step t+1 overlaps the WMMAs of
// step t; ping-pong LDS buffers give one __syncthreads per step.
// ---------------------------------------------------------------------------
template <int MODE>
__global__ __launch_bounds__(256)
void gemm_bf16_wmma(const void* __restrict__ Ap, const float* __restrict__ W,
                    const float* __restrict__ bias, void* __restrict__ Outp,
                    float scale) {
  constexpr int BM = 128, BN = 128, BK = 32;
  constexpr int KD = Emb, N = Emb;
  constexpr int NSTEP = KD / BK;              // 64
  __shared__ bf16 sA[2][BM][BK + 2];          // [buf][m][k]
  __shared__ bf16 sB[2][BN][BK + 2];          // [buf][n][k] (transposed)

  const int tid  = threadIdx.x;
  const int lane = tid & 31;
  const int wav  = tid >> 5;
  const int wm   = wav >> 1;          // 0..3  (M strip)
  const int wn   = wav & 1;           // 0..1  (N half)
  const int m0   = blockIdx.y * BM;
  const int n0   = blockIdx.x * BN;
  const int hi   = lane >> 4;         // lane group (0/1)
  const int lo   = lane & 15;

  const float* Af = (const float*)Ap;
  const bf16*  Ab = (const bf16*)Ap;

  // staging-thread coordinates
  const int akk = tid & 31, amb = tid >> 5;         // A: col, row-base
  const int bn  = tid & 127, bkb = (tid >> 7) * 16; // B: col, k-base

  auto loadA = [&](int k0, float* r) {
#pragma unroll
    for (int i = 0; i < 16; ++i) {
      const int m = amb + i * 8;
      r[i] = (MODE == 0) ? Af[(size_t)(m0 + m) * KD + (k0 + akk)]
                         : (float)Ab[(size_t)(m0 + m) * KD + (k0 + akk)];
    }
  };
  auto loadB = [&](int k0, float* r) {
#pragma unroll
    for (int i = 0; i < 16; ++i)
      r[i] = W[(size_t)(k0 + bkb + i) * N + (n0 + bn)];
  };
  auto storeTiles = [&](int buf, const float* ra, const float* rb) {
#pragma unroll
    for (int i = 0; i < 16; ++i) sA[buf][amb + i * 8][akk] = (bf16)ra[i];
#pragma unroll
    for (int i = 0; i < 16; ++i) sB[buf][bn][bkb + i] = (bf16)rb[i];
  };

  const v8f vzero = {0.f, 0.f, 0.f, 0.f, 0.f, 0.f, 0.f, 0.f};
  v8f acc[2][4];
#pragma unroll
  for (int i = 0; i < 2; ++i)
#pragma unroll
    for (int j = 0; j < 4; ++j) acc[i][j] = vzero;

  float ra[16], rb[16];
  loadA(0, ra);
  loadB(0, rb);
  storeTiles(0, ra, rb);
  __syncthreads();

  for (int t = 0; t < NSTEP; ++t) {
    const int buf = t & 1;
    const bool more = (t + 1) < NSTEP;
    if (more) {                       // issue next-tile global loads early
      loadA((t + 1) * BK, ra);
      loadB((t + 1) * BK, rb);
    }

    // --- fragments (contiguous k = 16*hi + j for A and B) + WMMA on current buffer ---
    v16bf afr[2], bfr[4];
#pragma unroll
    for (int mt = 0; mt < 2; ++mt) {
      const int row = wm * 32 + mt * 16 + lo;
#pragma unroll
      for (int j = 0; j < 16; ++j) {
        const int k = hi * 16 + j;                   // A: K 0..15 / 16..31
        afr[mt][j] = sA[buf][row][k];
      }
    }
#pragma unroll
    for (int nt = 0; nt < 4; ++nt) {
      const int n = wn * 64 + nt * 16 + lo;
#pragma unroll
      for (int j = 0; j < 16; ++j)
        bfr[nt][j] = sB[buf][n][hi * 16 + j];        // B: K 0..15 / 16..31
    }
#pragma unroll
    for (int mt = 0; mt < 2; ++mt)
#pragma unroll
      for (int nt = 0; nt < 4; ++nt)
        acc[mt][nt] = wmma_bf16(afr[mt], bfr[nt], acc[mt][nt]);

    if (more) {                       // fill the other buffer, one barrier
      storeTiles(buf ^ 1, ra, rb);
      __syncthreads();
    }
  }

  // --- epilogue: C layout VGPR r -> rows r (lanes 0-15) / r+8 (lanes 16-31) ---
#pragma unroll
  for (int mt = 0; mt < 2; ++mt)
#pragma unroll
    for (int nt = 0; nt < 4; ++nt)
#pragma unroll
      for (int r = 0; r < 8; ++r) {
        const int gm = m0 + wm * 32 + mt * 16 + r + hi * 8;
        const int gn = n0 + wn * 64 + nt * 16 + lo;
        float v = acc[mt][nt][r] + bias[gn];
        if (MODE == 0) {
          v *= scale;
          const int b = gm >> 11, s = gm & (Seq - 1);
          const int h = gn >> 7, d = gn & (Dh - 1);
          ((bf16*)Outp)[(((size_t)(b * Hn + h)) * Seq + s) * Dh + d] = (bf16)v;
        } else {
          ((float*)Outp)[(size_t)gm * N + gn] = v;
        }
      }
}

// ---------------------------------------------------------------------------
// Flash-attention core: one block per (b, h, 128-row Q tile); 8 waves, each
// wave owns 16 Q rows. Q fragments pinned in VGPRs; K/V tiles (64 kv rows)
// staged in LDS per iteration (+ global_prefetch_b8 look-ahead for the next
// tile); online softmax in f32; P re-shaped through a per-wave LDS tile into
// A-fragment layout for the PV WMMA. Fragment registers are explicitly
// double-buffered so ds_load issues a full stage ahead of the consuming WMMA.
// ---------------------------------------------------------------------------
__global__ __launch_bounds__(256)
void attn_fwd_wmma(const bf16* __restrict__ Qb, const bf16* __restrict__ Kb,
                   const bf16* __restrict__ Vb, const int* __restrict__ mask,
                   bf16* __restrict__ Xo) {
  constexpr int BQ = 128, BKV = 64;
  __shared__ alignas(32) bf16 sK[BKV][Dh + 16];    // [kv][d], 288B rows (32-mult)
  __shared__ alignas(32) bf16 sVt[Dh][BKV + 16];   // [d][kv], 160B rows (32-mult)
  __shared__ bf16 sP[8][16][BKV + 8];              // per-wave probability tile

  const int tid  = threadIdx.x;
  const int lane = tid & 31;
  const int wav  = tid >> 5;
  const int hi   = lane >> 4, lo = lane & 15;
  const int q0   = blockIdx.x * BQ;
  const int h    = blockIdx.y;
  const int b    = blockIdx.z;

  const bf16* Qh = Qb + ((size_t)(b * Hn + h)) * Seq * Dh;
  const bf16* Kh = Kb + ((size_t)(b * Hn + h)) * Seq * Dh;
  const bf16* Vh = Vb + ((size_t)(b * Hn + h)) * Seq * Dh;
  const int*  mb = mask + (size_t)b * Seq * Seq;

  // Q fragments: 16 rows x 128 d -> 4 A-fragments, loaded once from global.
  v16bf qf[4];
  {
    const size_t row = (size_t)(q0 + wav * 16 + lo);
#pragma unroll
    for (int kt = 0; kt < 4; ++kt)
#pragma unroll
      for (int j = 0; j < 16; ++j) {
        const int d = kt * 32 + hi * 16 + j;
        qf[kt][j] = Qh[row * Dh + d];
      }
  }

  const v8f vzero = {0.f, 0.f, 0.f, 0.f, 0.f, 0.f, 0.f, 0.f};
  v8f o[8];
#pragma unroll
  for (int dt = 0; dt < 8; ++dt) o[dt] = vzero;
  float mrow[8], lrow[8];
#pragma unroll
  for (int r = 0; r < 8; ++r) { mrow[r] = -3.0e38f; lrow[r] = 0.f; }

  // per-thread staging coordinates for K/V tiles
  const int srow = tid >> 2;            // 0..63
  const int sdb  = (tid & 3) * 32;      // 0,32,64,96

  for (int kv0 = 0; kv0 < Seq; kv0 += BKV) {
    __syncthreads();   // protect sK/sVt reuse
    // --- stage K tile (b128 vector copies) and V tile transposed ---
    {
      const int4* ks = (const int4*)(Kh + ((size_t)(kv0 + srow)) * Dh + sdb);
      int4* kd = (int4*)(&sK[srow][sdb]);
#pragma unroll
      for (int i = 0; i < 4; ++i) kd[i] = ks[i];

      union { int4 q[4]; bf16 hh[32]; } vbuf;
      const int4* vs = (const int4*)(Vh + ((size_t)(kv0 + srow)) * Dh + sdb);
#pragma unroll
      for (int i = 0; i < 4; ++i) vbuf.q[i] = vs[i];
#pragma unroll
      for (int i = 0; i < 32; ++i) sVt[sdb + i][srow] = vbuf.hh[i];

      // look-ahead: pull next K/V tile + next mask rows toward the caches
      if (kv0 + BKV < Seq) {
        __builtin_prefetch(Kh + ((size_t)(kv0 + BKV + srow)) * Dh + sdb, 0, 3);
        __builtin_prefetch(Vh + ((size_t)(kv0 + BKV + srow)) * Dh + sdb, 0, 3);
        __builtin_prefetch(mb + (size_t)(q0 + wav * 16 + lo) * Seq +
                               (kv0 + BKV), 0, 2);
      }
    }
    __syncthreads();

    // --- scores: S(16x64) = Q(16x128) x K^T, 4 k-steps x 4 n-tiles ---
    // K fragments double-buffered: loads for kt+1 issue under WMMAs of kt.
    v8f sc[4] = {vzero, vzero, vzero, vzero};
    v16bf kfr[2][4];
#pragma unroll
    for (int nt = 0; nt < 4; ++nt)
      kfr[0][nt] = *(const v16bf*)&sK[nt * 16 + lo][hi * 16];
#pragma unroll
    for (int kt = 0; kt < 4; ++kt) {
      const int cur = kt & 1;
      if (kt < 3) {
#pragma unroll
        for (int nt = 0; nt < 4; ++nt)
          kfr[cur ^ 1][nt] =
              *(const v16bf*)&sK[nt * 16 + lo][(kt + 1) * 32 + hi * 16];
      }
#pragma unroll
      for (int nt = 0; nt < 4; ++nt)
        sc[nt] = wmma_bf16(qf[kt], kfr[cur][nt], sc[nt]);
    }

    // --- additive mask bias + online softmax (row stats via shfl-xor<16) ---
#pragma unroll
    for (int r = 0; r < 8; ++r) {
      const int qg = q0 + wav * 16 + r + hi * 8;
      float mx = -3.0e38f;
#pragma unroll
      for (int nt = 0; nt < 4; ++nt) {
        const int kg = kv0 + nt * 16 + lo;
        const float sv =
            sc[nt][r] + ((mb[(size_t)qg * Seq + kg] > 0) ? 0.f : -1.0e10f);
        sc[nt][r] = sv;
        mx = fmaxf(mx, sv);
      }
#pragma unroll
      for (int off = 1; off < 16; off <<= 1)
        mx = fmaxf(mx, __shfl_xor(mx, off, 32));
      const float mn   = fmaxf(mrow[r], mx);
      const float corr = __expf(mrow[r] - mn);
      mrow[r] = mn;
      float rs = 0.f;
#pragma unroll
      for (int nt = 0; nt < 4; ++nt) {
        const float p = __expf(sc[nt][r] - mn);
        rs += p;
        sP[wav][r + hi * 8][nt * 16 + lo] = (bf16)p;
      }
#pragma unroll
      for (int off = 1; off < 16; off <<= 1) rs += __shfl_xor(rs, off, 32);
      lrow[r] = lrow[r] * corr + rs;
#pragma unroll
      for (int dt = 0; dt < 8; ++dt) o[dt][r] *= corr;
    }

    // --- PV: O(16x128) += P(16x64) x V(64x128); per-wave sP, DS in-order ---
    v16bf pf[2];
#pragma unroll
    for (int ks2 = 0; ks2 < 2; ++ks2)
#pragma unroll
      for (int j = 0; j < 16; ++j) {
        const int kv = ks2 * 32 + hi * 16 + j;
        pf[ks2][j] = sP[wav][lo][kv];
      }
    // V fragments double-buffered across the 16 (ks2, dt) steps.
    v16bf vfr[2];
    vfr[0] = *(const v16bf*)&sVt[lo][hi * 16];   // ks2=0, dt=0
#pragma unroll
    for (int step = 0; step < 16; ++step) {
      const int ks2 = step >> 3;
      const int dt  = step & 7;
      const int cur = step & 1;
      if (step < 15) {
        const int nks = (step + 1) >> 3;
        const int ndt = (step + 1) & 7;
        vfr[cur ^ 1] =
            *(const v16bf*)&sVt[ndt * 16 + lo][nks * 32 + hi * 16];
      }
      o[dt] = wmma_bf16(pf[ks2], vfr[cur], o[dt]);
    }
  }

  // --- normalize and store attention output as bf16 [B,S,H*D] ---
#pragma unroll
  for (int dt = 0; dt < 8; ++dt)
#pragma unroll
    for (int r = 0; r < 8; ++r) {
      const int s = q0 + wav * 16 + r + hi * 8;
      const int d = dt * 16 + lo;
      Xo[((size_t)b * Seq + s) * Emb + h * Dh + d] = (bf16)(o[dt][r] / lrow[r]);
    }
}

// ---------------------------------------------------------------------------
extern "C" void kernel_launch(void* const* d_in, const int* in_sizes, int n_in,
                              void* d_out, int out_size, void* d_ws,
                              size_t ws_size, hipStream_t stream) {
  const float* inputs_q  = (const float*)d_in[0];
  const float* inputs_kv = (const float*)d_in[1];
  const int*   mask      = (const int*)d_in[2];
  const float* wq = (const float*)d_in[3];
  const float* bq = (const float*)d_in[4];
  const float* wk = (const float*)d_in[5];
  const float* bk = (const float*)d_in[6];
  const float* wv = (const float*)d_in[7];
  const float* bv = (const float*)d_in[8];
  const float* wo = (const float*)d_in[9];
  const float* bo = (const float*)d_in[10];
  float* out = (float*)d_out;

  // Workspace: bf16 Q, K, V in [B,H,S,D] plus attention output [B,S,E]: 64 MB.
  const size_t nQKV = (size_t)Bsz * Hn * Seq * Dh;   // 8,388,608 elements
  bf16* Qb = (bf16*)d_ws;
  bf16* Kb = Qb + nQKV;
  bf16* Vb = Kb + nQKV;
  bf16* Xb = Vb + nQKV;

  const dim3 blk(256);
  const dim3 gg(Emb / 128, Mtok / 128);              // 16 x 32 blocks
  const float qscale = 0.088388347648318447f;        // 1/sqrt(D)

  gemm_bf16_wmma<0><<<gg, blk, 0, stream>>>(inputs_q,  wq, bq, Qb, qscale);
  gemm_bf16_wmma<0><<<gg, blk, 0, stream>>>(inputs_kv, wk, bk, Kb, 1.0f);
  gemm_bf16_wmma<0><<<gg, blk, 0, stream>>>(inputs_kv, wv, bv, Vb, 1.0f);
  attn_fwd_wmma<<<dim3(Seq / 128, Hn, Bsz), blk, 0, stream>>>(Qb, Kb, Vb, mask,
                                                              Xb);
  gemm_bf16_wmma<1><<<gg, blk, 0, stream>>>(Xb, wo, bo, out, 1.0f);

  (void)in_sizes; (void)n_in; (void)out_size; (void)ws_size;
}
